// TransformerBlock_18485539242042
// MI455X (gfx1250) — matrix operand order repaired and checked by reading, not compiled, not run on hardware
//
#include <hip/hip_runtime.h>

// ---------------------------------------------------------------------------
// CDNA5 (gfx1250) transformer block:
//   - bf16 WMMA (v_wmma_f32_16x16x32_bf16) for all GEMM-shaped work
//   - async global->LDS staging (global_load_async_to_lds_b128 + s_wait_asynccnt)
//   - block-cooperative causal flash attention (no S x S materialization)
// ---------------------------------------------------------------------------

typedef __attribute__((ext_vector_type(16))) __bf16 v16bf;
typedef __attribute__((ext_vector_type(8)))  __bf16 v8bf;
typedef __attribute__((ext_vector_type(8)))  float  v8f;

union BF16x16 { v16bf v; v8bf h[2]; };

__device__ __forceinline__ __bf16 f2bf(float f) {
  unsigned u = __builtin_bit_cast(unsigned, f);
  u += 0x7FFFu + ((u >> 16) & 1u);                 // round-to-nearest-even
  unsigned short hs = (unsigned short)(u >> 16);
  return __builtin_bit_cast(__bf16, hs);
}
__device__ __forceinline__ float bf2f(__bf16 b) {
  unsigned short hs = __builtin_bit_cast(unsigned short, b);
  unsigned u = ((unsigned)hs) << 16;
  return __builtin_bit_cast(float, u);
}
__device__ __forceinline__ v8f wmma_bf16(v16bf a, v16bf b, v8f c) {
  return __builtin_amdgcn_wmma_f32_16x16x32_bf16(false, a, false, b,
                                                 (short)0, c, false, false);
}

// Low 32 bits of a generic pointer into LDS == wave-relative LDS byte offset
// (flat->LDS aperture mapping truncates to addr[31:0], ISA 10.2).
__device__ __forceinline__ unsigned lds_off(const void* p) {
  return (unsigned)(size_t)p;
}
// Async DMA: memory -> LDS, 16B per lane, tracked by ASYNCcnt.
__device__ __forceinline__ void async_ld_b128(unsigned lds, const void* gaddr) {
  asm volatile("global_load_async_to_lds_b128 %0, %1, off"
               :: "v"(lds), "v"((unsigned long long)(size_t)gaddr) : "memory");
}
__device__ __forceinline__ void wait_async0() {
  asm volatile("s_wait_asynccnt 0x0" ::: "memory");
}

// ---------------------------------------------------------------------------
// Weight convert + transpose: W[K][N] f32 -> Wt[N][K] bf16 (B-operand layout)
// ---------------------------------------------------------------------------
__global__ void cvt_transpose_kernel(const float* __restrict__ in,
                                     __bf16* __restrict__ out, int K, int N) {
  long idx = (long)blockIdx.x * blockDim.x + threadIdx.x;
  if (idx >= (long)K * N) return;
  int kk = (int)(idx / N);
  int n  = (int)(idx % N);
  out[(size_t)n * K + kk] = f2bf(in[idx]);
}

// ---------------------------------------------------------------------------
// RMSNorm: f32 row -> bf16 row, one block per row of D=1024
// ---------------------------------------------------------------------------
__global__ __launch_bounds__(256)
void rmsnorm_kernel(const float* __restrict__ x, const float* __restrict__ g,
                    __bf16* __restrict__ out, int D) {
  __shared__ float red[256];
  const int row = blockIdx.x;
  const float* xr = x + (size_t)row * D;
  float s = 0.f;
  for (int i = threadIdx.x; i < D; i += 256) { float v = xr[i]; s += v * v; }
  red[threadIdx.x] = s;
  __syncthreads();
  for (int st = 128; st > 0; st >>= 1) {
    if (threadIdx.x < st) red[threadIdx.x] += red[threadIdx.x + st];
    __syncthreads();
  }
  const float inv = rsqrtf(red[0] / (float)D + 1e-5f);
  for (int i = threadIdx.x; i < D; i += 256)
    out[(size_t)row * D + i] = f2bf(xr[i] * inv * g[i]);
}

// ---------------------------------------------------------------------------
// GEMM: C[M][N] = A[M][K](bf16) * Bt[N][K](bf16)^T
// 256 thr = 8 waves; block tile 64x256; wave tile 32x64 (8 WMMA / K-step);
// K-step 32; tiles staged with async global->LDS DMA.
// EPI==0: store bf16. EPI==1: Cf = resid + acc (f32).
// ---------------------------------------------------------------------------
template <int EPI>
__global__ __launch_bounds__(256)
void gemm_bf16_kernel(const __bf16* __restrict__ A,
                      const __bf16* __restrict__ Bt,
                      __bf16* __restrict__ Cb,
                      float* __restrict__ Cf,
                      const float* __restrict__ resid,
                      int M, int N, int K) {
  constexpr int BM = 64, BN = 256, BK = 32;
  __shared__ __align__(16) __bf16 As[BM][BK + 8];   // row stride 80B (16B mult)
  __shared__ __align__(16) __bf16 Bs[BN][BK + 8];

  const int t      = threadIdx.x;
  const int lane   = t & 31;
  const int wave   = t >> 5;
  const int wm     = wave >> 2;              // 0..1 -> 32 rows each
  const int wn     = wave & 3;               // 0..3 -> 64 cols each
  const int mBase  = blockIdx.y * BM;
  const int nBase  = blockIdx.x * BN;
  const int l16    = lane & 15;
  const int halfHi = lane >> 4;              // 0|1

  const v8f vzero = {};
  v8f acc[2][4];
  #pragma unroll
  for (int i = 0; i < 2; i++)
    #pragma unroll
    for (int j = 0; j < 4; j++) acc[i][j] = vzero;

  for (int kb = 0; kb < K; kb += BK) {
    {   // A tile: 64x32 halves = 256 x 16B chunks, one per thread (async DMA)
      int row = t >> 2, ck = (t & 3) << 3;
      async_ld_b128(lds_off(&As[row][ck]),
                    A + (size_t)(mBase + row) * K + kb + ck);
    }
    #pragma unroll
    for (int i = 0; i < 4; i++) {   // B tile: 256x32 halves = 1024 chunks
      int cid = t + i * 256;
      int row = cid >> 2, ck = (cid & 3) << 3;
      async_ld_b128(lds_off(&Bs[row][ck]),
                    Bt + (size_t)(nBase + row) * K + kb + ck);
    }
    if (kb + BK < K)   // cache hint for next A tile (global_prefetch_b8)
      __builtin_prefetch(A + (size_t)(mBase + (t >> 2)) * K + kb + BK, 0, 1);
    wait_async0();
    __syncthreads();

    BF16x16 aF[2], bF[4];
    #pragma unroll
    for (int i = 0; i < 2; i++) {        // A frag: 16-bit A layout
      int row = wm * 32 + i * 16 + l16;
      int k0  = halfHi * 8;
      aF[i].h[0] = *(const v8bf*)&As[row][k0];        // K k0..k0+7
      aF[i].h[1] = *(const v8bf*)&As[row][k0 + 16];   // K k0+16..k0+23
    }
    #pragma unroll
    for (int j = 0; j < 4; j++) {        // B frag: 16-bit B layout (same K order as A)
      int col = wn * 64 + j * 16 + l16;
      int k0  = halfHi * 8;
      bF[j].h[0] = *(const v8bf*)&Bs[col][k0];
      bF[j].h[1] = *(const v8bf*)&Bs[col][k0 + 16];
    }
    #pragma unroll
    for (int i = 0; i < 2; i++)
      #pragma unroll
      for (int j = 0; j < 4; j++)
        acc[i][j] = wmma_bf16(aF[i].v, bF[j].v, acc[i][j]);
    __syncthreads();
  }

  #pragma unroll
  for (int i = 0; i < 2; i++)
    #pragma unroll
    for (int j = 0; j < 4; j++)
      #pragma unroll
      for (int r = 0; r < 8; r++) {
        int m = mBase + wm * 32 + i * 16 + r + halfHi * 8;
        int n = nBase + wn * 64 + j * 16 + l16;
        size_t idx = (size_t)m * N + n;
        float val = acc[i][j][r];
        if constexpr (EPI == 0) Cb[idx] = f2bf(val);
        else                    Cf[idx] = resid[idx] + val;
      }
}

// ---------------------------------------------------------------------------
// RoPE in-place on bf16 [B][S][H][64]; one thread per (even,odd) pair
// ---------------------------------------------------------------------------
__global__ void rope_kernel(__bf16* __restrict__ x, int total) {
  int idx = blockIdx.x * blockDim.x + threadIdx.x;
  if (idx >= total) return;
  int j    = idx & 31;                    // pair index 0..31
  int rest = idx >> 5;                    // (b*S + s)*H + h
  int s    = (rest >> 4) & 2047;          // H=16, S=2048
  float inv = powf(10000.0f, -(float)(2 * j) / 64.0f);
  float ang = (float)s * inv;
  float c = cosf(ang), sn = sinf(ang);
  __bf16* p = x + ((size_t)idx << 1);
  float e = bf2f(p[0]), o = bf2f(p[1]);
  p[0] = f2bf(e * c - o * sn);
  p[1] = f2bf(e * sn + o * c);
}

// ---------------------------------------------------------------------------
// Flash attention, block-cooperative:
//   block = 4 waves = 64 query rows of one (b,h); K/V tiles (32 keys) staged
//   once per block (K via async DMA, V transposed for the P@V B-operand).
//   Waves past their causal bound compute fully-masked tiles (p == 0; M,L
//   provably unchanged) so barriers stay uniform and EXEC stays all-ones.
// ---------------------------------------------------------------------------
__global__ __launch_bounds__(128)
void flash_attn_kernel(const __bf16* __restrict__ q,
                       const __bf16* __restrict__ k,
                       const __bf16* __restrict__ v,
                       __bf16* __restrict__ o) {
  constexpr int S = 2048, H = 16, DH = 64, QT = 16, KT = 32;
  __shared__ __align__(16) __bf16 Ks[KT][DH];      // [key][d]  (B for Q@K^T)
  __shared__ __align__(16) __bf16 Vs[DH][KT];      // [d][key]  (B for P@V)
  __shared__ __align__(16) __bf16 Ps[4][QT][KT];   // per-wave P staging

  const int t      = threadIdx.x;
  const int lane   = t & 31;
  const int wv     = t >> 5;                 // 0..3
  const int l16    = lane & 15;
  const int halfHi = lane >> 4;

  // block -> (b, h, 64-row q stripe); wave -> 16-row q tile
  int gb  = blockIdx.x;
  int qs  = gb & 31;                  // S/64 = 32 stripes
  int h   = (gb >> 5) & 15;
  int b   = gb >> 9;
  int qb0 = qs * 64;
  int qm  = qb0 + wv * QT;

  // Q fragments (persist across K loop)
  BF16x16 qA[2];
  {
    const __bf16* qrow = q + ((size_t)((b * S + qm + l16) * H + h)) * DH;
    #pragma unroll
    for (int h2 = 0; h2 < 2; h2++) {
      int d0 = h2 * 32 + halfHi * 8;
      qA[h2].h[0] = *(const v8bf*)(qrow + d0);
      qA[h2].h[1] = *(const v8bf*)(qrow + d0 + 16);
    }
  }

  const v8f vzero = {};
  v8f ctx[4];
  #pragma unroll
  for (int j = 0; j < 4; j++) ctx[j] = vzero;
  float Mx[8], Lx[8];
  #pragma unroll
  for (int r = 0; r < 8; r++) { Mx[r] = -3.0e38f; Lx[r] = 0.f; }
  const float sm_scale = 0.125f;     // 1/sqrt(64)

  for (int kb = 0; kb <= qb0 + 63; kb += KT) {
    // --- stage K tile (async DMA): 32 keys x 32 halves (d 0..31) = 128 x 16B chunks --
    {
      int row = t >> 2, ck = (t & 3) << 3;
      async_ld_b128(lds_off(&Ks[row][ck]),
                    k + ((size_t)((b * S + kb + row) * H + h)) * DH + ck);
    }
    // --- stage V tile transposed: thread -> (key row, 16-d segment) --------
    {
      int row = t >> 2, d0 = (t & 3) << 4;
      const __bf16* vrow =
          v + ((size_t)((b * S + kb + row) * H + h)) * DH + d0;
      v8bf v0 = *(const v8bf*)(vrow);
      v8bf v1 = *(const v8bf*)(vrow + 8);
      #pragma unroll
      for (int e = 0; e < 8; e++) {
        Vs[d0 + e][row]     = v0[e];
        Vs[d0 + 8 + e][row] = v1[e];
      }
    }
    wait_async0();
    __syncthreads();

    // --- scores: 16x32 via two 16x16 WMMA chains over dh=64 ----------------
    v8f sc[2];
    #pragma unroll
    for (int ts = 0; ts < 2; ts++) {
      int key = ts * 16 + l16;
      int d0  = halfHi * 8;
      BF16x16 kB0, kB1;
      kB0.h[0] = *(const v8bf*)&Ks[key][d0];
      kB0.h[1] = *(const v8bf*)&Ks[key][d0 + 16];
      kB1.h[0] = *(const v8bf*)&Ks[key][32 + d0];
      kB1.h[1] = *(const v8bf*)&Ks[key][32 + d0 + 16];
      v8f c = vzero;
      c = wmma_bf16(qA[0].v, kB0.v, c);
      c = wmma_bf16(qA[1].v, kB1.v, c);
      sc[ts] = c;
    }
    // --- scale + causal mask (arithmetic, no divergence) -------------------
    #pragma unroll
    for (int ts = 0; ts < 2; ts++)
      #pragma unroll
      for (int r = 0; r < 8; r++) {
        int mrow = qm + r + halfHi * 8;
        int key  = kb + ts * 16 + l16;
        float sv = sc[ts][r] * sm_scale;
        sc[ts][r] = (key <= mrow) ? sv : -3.0e38f;
      }
    // --- online softmax: row stats live in 16-lane halves ------------------
    float rescale[8];
    #pragma unroll
    for (int r = 0; r < 8; r++) {
      float tm = fmaxf(sc[0][r], sc[1][r]);
      tm = fmaxf(tm, __shfl_xor(tm, 1, 32));
      tm = fmaxf(tm, __shfl_xor(tm, 2, 32));
      tm = fmaxf(tm, __shfl_xor(tm, 4, 32));
      tm = fmaxf(tm, __shfl_xor(tm, 8, 32));
      float Mn = fmaxf(Mx[r], tm);
      float so = __expf(Mx[r] - Mn);
      float p0 = __expf(sc[0][r] - Mn);
      float p1 = __expf(sc[1][r] - Mn);
      sc[0][r] = p0; sc[1][r] = p1;
      float rs = p0 + p1;
      rs += __shfl_xor(rs, 1, 32);
      rs += __shfl_xor(rs, 2, 32);
      rs += __shfl_xor(rs, 4, 32);
      rs += __shfl_xor(rs, 8, 32);
      Lx[r] = Lx[r] * so + rs;
      Mx[r] = Mn;
      rescale[r] = so;
    }
    // --- P: C-layout -> A-layout via per-wave LDS transpose ----------------
    #pragma unroll
    for (int ts = 0; ts < 2; ts++)
      #pragma unroll
      for (int r = 0; r < 8; r++)
        Ps[wv][r + halfHi * 8][ts * 16 + l16] = f2bf(sc[ts][r]);
    __builtin_amdgcn_wave_barrier();   // per-wave LDS region, DS in-order
    BF16x16 pA;
    {
      int k0 = halfHi * 8;
      pA.h[0] = *(const v8bf*)&Ps[wv][l16][k0];
      pA.h[1] = *(const v8bf*)&Ps[wv][l16][k0 + 16];
    }
    // --- ctx = ctx*rescale + P @ V (4 dh-subtiles) -------------------------
    #pragma unroll
    for (int j = 0; j < 4; j++) {
      #pragma unroll
      for (int r = 0; r < 8; r++) ctx[j][r] *= rescale[r];
      int d  = j * 16 + l16;
      int k0 = halfHi * 8;
      BF16x16 vB;
      vB.h[0] = *(const v8bf*)&Vs[d][k0];
      vB.h[1] = *(const v8bf*)&Vs[d][k0 + 16];
      ctx[j] = wmma_bf16(pA.v, vB.v, ctx[j]);
    }
    __syncthreads();   // all waves done with Ks/Vs before restaging
  }

  // --- finalize: divide by row sum, store bf16 [b][s][h][dh] ---------------
  #pragma unroll
  for (int j = 0; j < 4; j++)
    #pragma unroll
    for (int r = 0; r < 8; r++) {
      int mrow = qm + r + halfHi * 8;
      float val = ctx[j][r] / Lx[r];
      o[((size_t)((b * S + mrow) * H + h)) * DH + j * 16 + l16] = f2bf(val);
    }
}

// ---------------------------------------------------------------------------
// SwiGLU: gate <- silu(gate) * up   (bf16 in-place)
// ---------------------------------------------------------------------------
__global__ void swiglu_kernel(__bf16* __restrict__ gate,
                              const __bf16* __restrict__ up, long n) {
  long i = (long)blockIdx.x * blockDim.x + threadIdx.x;
  if (i >= n) return;
  float g = bf2f(gate[i]);
  float u = bf2f(up[i]);
  float si = g / (1.0f + __expf(-g));
  gate[i] = f2bf(si * u);
}

// ---------------------------------------------------------------------------
// Host orchestration
// ---------------------------------------------------------------------------
extern "C" void kernel_launch(void* const* d_in, const int* in_sizes, int n_in,
                              void* d_out, int out_size, void* d_ws, size_t ws_size,
                              hipStream_t stream) {
  (void)in_sizes; (void)n_in; (void)out_size; (void)ws_size;
  const float* x  = (const float*)d_in[0];
  const float* Wq = (const float*)d_in[1];
  const float* Wk = (const float*)d_in[2];
  const float* Wv = (const float*)d_in[3];
  const float* Wo = (const float*)d_in[4];
  const float* W1 = (const float*)d_in[5];
  const float* W2 = (const float*)d_in[6];
  const float* W3 = (const float*)d_in[7];
  const float* g1 = (const float*)d_in[8];
  const float* g2 = (const float*)d_in[9];
  float* out = (float*)d_out;

  constexpr int B = 2, S = 2048, D = 1024, H = 16, F = 4096;
  const int M = B * S;                    // 4096 token rows

  // workspace layout (~160 MB, 256B-aligned slots)
  char* ws = (char*)d_ws;
  size_t off = 0;
  auto alloc = [&](size_t bytes) -> void* {
    void* p = ws + off;
    off += (bytes + 255) & ~(size_t)255;
    return p;
  };
  __bf16* Wqt  = (__bf16*)alloc((size_t)D * D * 2);
  __bf16* Wkt  = (__bf16*)alloc((size_t)D * D * 2);
  __bf16* Wvt  = (__bf16*)alloc((size_t)D * D * 2);
  __bf16* Wot  = (__bf16*)alloc((size_t)D * D * 2);
  __bf16* W1t  = (__bf16*)alloc((size_t)D * F * 2);
  __bf16* W3t  = (__bf16*)alloc((size_t)D * F * 2);
  __bf16* W2t  = (__bf16*)alloc((size_t)F * D * 2);
  __bf16* xn   = (__bf16*)alloc((size_t)M * D * 2);
  __bf16* qb   = (__bf16*)alloc((size_t)M * D * 2);
  __bf16* kbuf = (__bf16*)alloc((size_t)M * D * 2);
  __bf16* vb   = (__bf16*)alloc((size_t)M * D * 2);
  __bf16* ctx  = (__bf16*)alloc((size_t)M * D * 2);
  float*  x1   = (float*) alloc((size_t)M * D * 4);
  __bf16* xn2  = (__bf16*)alloc((size_t)M * D * 2);
  __bf16* gate = (__bf16*)alloc((size_t)M * F * 2);
  __bf16* up   = (__bf16*)alloc((size_t)M * F * 2);

  auto cvt = [&](const float* w, __bf16* wt, int K, int N) {
    long n = (long)K * N;
    cvt_transpose_kernel<<<dim3((unsigned)((n + 255) / 256)), dim3(256), 0,
                           stream>>>(w, wt, K, N);
  };
  auto gemm0 = [&](const __bf16* A, const __bf16* Bt, __bf16* Cb,
                   int Mm, int Nn, int Kk) {
    gemm_bf16_kernel<0><<<dim3(Nn / 256, Mm / 64), dim3(256), 0, stream>>>(
        A, Bt, Cb, nullptr, nullptr, Mm, Nn, Kk);
  };
  auto gemm1 = [&](const __bf16* A, const __bf16* Bt, float* Cf,
                   const float* resid, int Mm, int Nn, int Kk) {
    gemm_bf16_kernel<1><<<dim3(Nn / 256, Mm / 64), dim3(256), 0, stream>>>(
        A, Bt, nullptr, Cf, resid, Mm, Nn, Kk);
  };

  // 1) weights -> bf16 [N][K]
  cvt(Wq, Wqt, D, D);  cvt(Wk, Wkt, D, D);  cvt(Wv, Wvt, D, D);
  cvt(Wo, Wot, D, D);  cvt(W1, W1t, D, F);  cvt(W3, W3t, D, F);
  cvt(W2, W2t, F, D);

  // 2) xn = rmsnorm(x, g1)
  rmsnorm_kernel<<<dim3(M), dim3(256), 0, stream>>>(x, g1, xn, D);

  // 3) q,k,v projections (WMMA GEMM)
  gemm0(xn, Wqt, qb,   M, D, D);
  gemm0(xn, Wkt, kbuf, M, D, D);
  gemm0(xn, Wvt, vb,   M, D, D);

  // 4) RoPE on q and k (in-place)
  {
    int pairs = M * H * 32;   // B*S*H*(dh/2)
    rope_kernel<<<dim3(pairs / 256), dim3(256), 0, stream>>>(qb,   pairs);
    rope_kernel<<<dim3(pairs / 256), dim3(256), 0, stream>>>(kbuf, pairs);
  }

  // 5) flash attention -> ctx  (block = 64-row q stripe of one (b,h))
  {
    int nBlocks = B * H * (S / 64);       // 1024
    flash_attn_kernel<<<dim3(nBlocks), dim3(128), 0, stream>>>(qb, kbuf, vb,
                                                               ctx);
  }

  // 6) x1 = x + ctx @ Wo
  gemm1(ctx, Wot, x1, x, M, D, D);

  // 7) xn2 = rmsnorm(x1, g2)
  rmsnorm_kernel<<<dim3(M), dim3(256), 0, stream>>>(x1, g2, xn2, D);

  // 8) gate = xn2@W1 ; up = xn2@W3
  gemm0(xn2, W1t, gate, M, F, D);
  gemm0(xn2, W3t, up,   M, F, D);

  // 9) gate <- silu(gate)*up
  {
    long n = (long)M * F;
    swiglu_kernel<<<dim3((unsigned)(n / 256)), dim3(256), 0, stream>>>(gate,
                                                                       up, n);
  }

  // 10) out = x1 + gate @ W2
  gemm1(gate, W2t, out, x1, M, D, F);
}
